// MultiHeadAttention_28690381537369
// MI455X (gfx1250) — matrix operand order repaired and checked by reading, not compiled, not run on hardware
//
#include <hip/hip_runtime.h>
#include <math.h>
#include <type_traits>

// Causal MHA: B=4, L=2048, D=1024, H=16, Dh=64. fp32 in/out, bf16 WMMA compute.
// cvt(x) + cvt/transpose(W*) once -> bf16; QKV GEMMs emit bf16 (Q pre-scaled
// by log2(e)/sqrt(Dh) so softmax runs in exp2 domain); flash attention
// (bf16 in/out, f32 online softmax); output GEMM emits fp32.

typedef __attribute__((ext_vector_type(16))) __bf16        v16bf;
typedef __attribute__((ext_vector_type(8)))  float         v8f;
typedef __attribute__((ext_vector_type(8)))  unsigned int  v8u;

#define BDIM 1024
#define LLEN 2048
#define NB   4
#define NH   16
#define DHH  64
#define MROWS (NB * LLEN)   // 8192

__device__ __forceinline__ unsigned short f2bf(float f) {
    unsigned int u = __builtin_bit_cast(unsigned int, f);
    unsigned int r = u + 0x7FFFu + ((u >> 16) & 1u);
    return (unsigned short)(r >> 16);
}
__device__ __forceinline__ unsigned int f2bf_pk(float lo, float hi) {
#if __has_builtin(__builtin_amdgcn_cvt_pk_bf16_f32)
    typedef __attribute__((ext_vector_type(2))) __bf16 v2bf;
    v2bf v = __builtin_amdgcn_cvt_pk_bf16_f32(lo, hi);
    return __builtin_bit_cast(unsigned int, v);
#else
    return (unsigned int)f2bf(lo) | ((unsigned int)f2bf(hi) << 16);
#endif
}
__device__ __forceinline__ v16bf bf_cast(v8u x) { return __builtin_bit_cast(v16bf, x); }

__device__ __forceinline__ float fast_exp2(float x) {
#if __has_builtin(__builtin_amdgcn_exp2f)
    return __builtin_amdgcn_exp2f(x);
#else
    return exp2f(x);
#endif
}
__device__ __forceinline__ float fast_rcp(float x) {
#if __has_builtin(__builtin_amdgcn_rcpf)
    return __builtin_amdgcn_rcpf(x);
#else
    return 1.0f / x;
#endif
}

// xor-butterfly lane exchange via ds_swizzle (group-of-32: and=0x1f, or=0, xor=MASK)
template <int MASK>
__device__ __forceinline__ float swz_xor(float x) {
    return __builtin_bit_cast(float,
        __builtin_amdgcn_ds_swizzle(__builtin_bit_cast(int, x), (MASK << 10) | 0x1F));
}
__device__ __forceinline__ float redmax16(float x) {
    x = fmaxf(x, swz_xor<1>(x)); x = fmaxf(x, swz_xor<2>(x));
    x = fmaxf(x, swz_xor<4>(x)); x = fmaxf(x, swz_xor<8>(x));
    return x;
}
__device__ __forceinline__ float redsum16(float x) {
    x += swz_xor<1>(x); x += swz_xor<2>(x);
    x += swz_xor<4>(x); x += swz_xor<8>(x);
    return x;
}

// ---------------------------------------------------------------------------
// One-time converts
// ---------------------------------------------------------------------------
__global__ __launch_bounds__(256) void cvt_bf16(
    const float* __restrict__ in, unsigned short* __restrict__ out)
{
    size_t i = ((size_t)blockIdx.x * 256 + threadIdx.x) * 4;
    float4 f = *(const float4*)(in + i);
    uint2 o;
    o.x = f2bf_pk(f.x, f.y);
    o.y = f2bf_pk(f.z, f.w);
    *(uint2*)(out + i) = o;
}

__global__ __launch_bounds__(256) void cvt_transpose_bf16(
    const float* __restrict__ W, unsigned short* __restrict__ Wt)
{
    __shared__ float tile[32][33];
    const int nb = blockIdx.x * 32, kb = blockIdx.y * 32;
    const int tx = threadIdx.x & 31, ty = threadIdx.x >> 5;
    #pragma unroll
    for (int j = 0; j < 32; j += 8)
        tile[ty + j][tx] = W[(size_t)(kb + ty + j) * BDIM + nb + tx];
    __syncthreads();
    #pragma unroll
    for (int j = 0; j < 32; j += 8)
        Wt[(size_t)(nb + ty + j) * BDIM + kb + tx] = f2bf(tile[tx][ty + j]);
}

// ---------------------------------------------------------------------------
// GEMM: Y[M,N] = X[M,K] @ Wt[N,K]^T + bias, bf16 operands, f32 accum.
// M=8192, N=K=1024 fixed. Workgroup tile 128x128, 8 waves 4x2, wave 32x64.
// ---------------------------------------------------------------------------
template <bool OUT_BF16>
__global__ __launch_bounds__(256) void gemm_bf16(
    const unsigned short* __restrict__ X,
    const unsigned short* __restrict__ Wt,
    const float* __restrict__ bias,
    void* __restrict__ Yv, float oscale)
{
    __shared__ unsigned short As[128 * 40];
    __shared__ unsigned short Bs[128 * 40];

    const int tid  = threadIdx.x;
    const int lane = tid & 31, wave = tid >> 5;
    const int half = lane >> 4, lm = lane & 15;
    const int wm = wave >> 1, wn = wave & 1;
    const int bm = blockIdx.x * 128, bn = blockIdx.y * 128;

    v8f acc[2][4] = {};
    for (int kk = 0; kk < BDIM; kk += 32) {
        #pragma unroll
        for (int i = 0; i < 2; ++i) {
            int e = (tid + 256 * i) * 8;
            int m = e >> 5, k = e & 31;
            *(uint4*)&As[m * 40 + k] =
                *(const uint4*)&X[(size_t)(bm + m) * BDIM + kk + k];
            *(uint4*)&Bs[m * 40 + k] =
                *(const uint4*)&Wt[(size_t)(bn + m) * BDIM + kk + k];
        }
        __syncthreads();

        v16bf a[2], b[4];
        #pragma unroll
        for (int mt = 0; mt < 2; ++mt) {
            int m = wm * 32 + mt * 16 + lm;
            v8u r;
            #pragma unroll
            for (int v = 0; v < 8; ++v) {
                int kf = (v < 4) ? (half * 8 + 2 * v) : (16 + half * 8 + 2 * (v - 4));
                r[v] = *(const unsigned int*)&As[m * 40 + kf];
            }
            a[mt] = bf_cast(r);
        }
        #pragma unroll
        for (int nt = 0; nt < 4; ++nt) {
            int n = wn * 64 + nt * 16 + lm;
            v8u r;
            #pragma unroll
            for (int v = 0; v < 8; ++v) {
                int kf = (v < 4) ? (half * 8 + 2 * v) : (16 + half * 8 + 2 * (v - 4));
                r[v] = *(const unsigned int*)&Bs[n * 40 + kf];
            }
            b[nt] = bf_cast(r);
        }
        #pragma unroll
        for (int mt = 0; mt < 2; ++mt)
            #pragma unroll
            for (int nt = 0; nt < 4; ++nt)
                acc[mt][nt] = __builtin_amdgcn_wmma_f32_16x16x32_bf16(
                    false, a[mt], false, b[nt], (short)0, acc[mt][nt], false, false);
        __syncthreads();
    }

    #pragma unroll
    for (int nt = 0; nt < 4; ++nt) {
        int n = bn + wn * 64 + nt * 16 + lm;
        float bv = bias[n];
        #pragma unroll
        for (int mt = 0; mt < 2; ++mt) {
            int m0 = bm + wm * 32 + mt * 16 + 8 * half;
            if (OUT_BF16) {
                unsigned short* yp = (unsigned short*)Yv + (size_t)m0 * BDIM + n;
                #pragma unroll
                for (int r = 0; r < 8; ++r)
                    yp[r * BDIM] = f2bf((acc[mt][nt][r] + bv) * oscale);
            } else {
                float* yp = (float*)Yv + (size_t)m0 * BDIM + n;
                #pragma unroll
                for (int r = 0; r < 8; ++r)
                    yp[r * BDIM] = (acc[mt][nt][r] + bv) * oscale;
            }
        }
    }
}

// ---------------------------------------------------------------------------
// Flash attention (causal). Grid (L/128, B*H); 8 waves, each a 16-row strip.
// Diagonal sub-block peeled into a separate template instantiation so the
// steady-state path carries zero masking code. Softmax in exp2 domain.
// ---------------------------------------------------------------------------
__global__ __launch_bounds__(256) void flash_attn_bf16(
    const unsigned short* __restrict__ Q, const unsigned short* __restrict__ Kg,
    const unsigned short* __restrict__ Vg, unsigned short* __restrict__ Og)
{
    __shared__ unsigned short Ks[128 * 72];   // [key][feat], pad 72
    __shared__ unsigned short Vs[64 * 132];   // [feat][key], pad 132
    __shared__ unsigned short Ps[8][16 * 34]; // per-wave P re-layout buffer

    const int tid  = threadIdx.x;
    const int lane = tid & 31, wave = tid >> 5;
    const int half = lane >> 4, lm = lane & 15;
    const int qblock = blockIdx.x;
    const int bh = blockIdx.y;
    const int b = bh / NH, h = bh % NH;
    const int qb = qblock * 128 + wave * 16;
    const int dks = wave >> 1;                 // diagonal sub-block index
    const size_t headoff = (size_t)b * LLEN * BDIM + (size_t)h * DHH;

    // Q fragments (pre-scaled by log2(e)/sqrt(Dh) in the Q GEMM)
    v16bf a0, a1;
    {
        v8u r0, r1;
        const unsigned short* qp = Q + headoff + (size_t)(qb + lm) * BDIM;
        #pragma unroll
        for (int v = 0; v < 8; ++v) {
            int kf = (v < 4) ? (half * 8 + 2 * v) : (16 + half * 8 + 2 * (v - 4));
            r0[v] = *(const unsigned int*)&qp[kf];
            r1[v] = *(const unsigned int*)&qp[32 + kf];
        }
        a0 = bf_cast(r0); a1 = bf_cast(r1);
    }

    float rowmax[8], rowsum[8];
    #pragma unroll
    for (int r = 0; r < 8; ++r) { rowmax[r] = -__builtin_inff(); rowsum[r] = 0.f; }
    v8f O[4] = {};

    int kbb = 0;   // current macroblock key base (captured by reference below)

    // one 32-key sub-block: QK^T, online softmax (exp2 domain), P@V
    auto process = [&](int ksub, auto MASKED_C) {
        constexpr bool MASKED = decltype(MASKED_C)::value;
        const int kbase = kbb + ksub * 32;

        v8f S[2] = {};
        #pragma unroll
        for (int kt = 0; kt < 2; ++kt) {
            int key = ksub * 32 + kt * 16 + lm;
            v8u rb0, rb1;
            #pragma unroll
            for (int v = 0; v < 8; ++v) {
                int f0 = (v < 4) ? (half * 8 + 2 * v) : (16 + half * 8 + 2 * (v - 4));
                rb0[v] = *(const unsigned int*)&Ks[key * 72 + f0];
                rb1[v] = *(const unsigned int*)&Ks[key * 72 + 32 + f0];
            }
            v8f s = __builtin_amdgcn_wmma_f32_16x16x32_bf16(
                false, a0, false, bf_cast(rb0), (short)0, S[kt], false, false);
            S[kt] = __builtin_amdgcn_wmma_f32_16x16x32_bf16(
                false, a1, false, bf_cast(rb1), (short)0, s, false, false);
        }

        float p0[8], p1[8];
        #pragma unroll
        for (int r = 0; r < 8; ++r) {
            float s0 = S[0][r], s1 = S[1][r];
            if (MASKED) {
                int qg = qb + r + 8 * half;
                s0 = (kbase + lm      > qg) ? -__builtin_inff() : s0;
                s1 = (kbase + 16 + lm > qg) ? -__builtin_inff() : s1;
            }
            float mx = redmax16(fmaxf(s0, s1));
            float newmax = fmaxf(rowmax[r], mx);
            float scale  = fast_exp2(rowmax[r] - newmax);
            float e0 = fast_exp2(s0 - newmax);
            float e1 = fast_exp2(s1 - newmax);
            rowsum[r] = rowsum[r] * scale + e0 + e1;   // per-lane partial
            rowmax[r] = newmax;
            p0[r] = e0; p1[r] = e1;
            #pragma unroll
            for (int t = 0; t < 4; ++t) O[t][r] *= scale;
        }

        // P: C layout -> A layout via wave-private LDS (in-order DS ops)
        #pragma unroll
        for (int r = 0; r < 8; ++r) {
            int m = r + 8 * half;
            Ps[wave][m * 34 + lm]      = f2bf(p0[r]);
            Ps[wave][m * 34 + 16 + lm] = f2bf(p1[r]);
        }
        __asm volatile("s_wait_dscnt 0" ::: "memory");
        v8u rp;
        #pragma unroll
        for (int v = 0; v < 8; ++v) {
            int kf = (v < 4) ? (half * 8 + 2 * v) : (16 + half * 8 + 2 * (v - 4));
            rp[v] = *(const unsigned int*)&Ps[wave][lm * 34 + kf];
        }
        v16bf pA = bf_cast(rp);

        #pragma unroll
        for (int t = 0; t < 4; ++t) {
            v8u rv;
            #pragma unroll
            for (int v = 0; v < 8; ++v) {
                int key = ksub * 32 + ((v < 4) ? (half * 8 + 2 * v) : (16 + half * 8 + 2 * (v - 4)));
                rv[v] = *(const unsigned int*)&Vs[(t * 16 + lm) * 132 + key];
            }
            O[t] = __builtin_amdgcn_wmma_f32_16x16x32_bf16(
                false, pA, false, bf_cast(rv), (short)0, O[t], false, false);
        }
    };

    for (int kb = 0; kb <= qblock; ++kb) {
        kbb = kb * 128;
        // stage K [key][feat]
        #pragma unroll
        for (int i = 0; i < 4; ++i) {
            int e = (tid + 256 * i) * 8;
            int key = e >> 6, feat = e & 63;
            *(uint4*)&Ks[key * 72 + feat] =
                *(const uint4*)&Kg[headoff + (size_t)(kbb + key) * BDIM + feat];
        }
        // stage V transposed [feat][key]: key-pair pack -> ds_store_b32
        #pragma unroll
        for (int i = 0; i < 2; ++i) {
            int p = tid + 256 * i;
            int key0 = (p >> 3) * 2, fg = (p & 7) * 8;
            const unsigned short* vp = Vg + headoff + (size_t)(kbb + key0) * BDIM + fg;
            uint4 q0 = *(const uint4*)vp;
            uint4 q1 = *(const uint4*)(vp + BDIM);
            unsigned int w0[4] = {q0.x, q0.y, q0.z, q0.w};
            unsigned int w1[4] = {q1.x, q1.y, q1.z, q1.w};
            #pragma unroll
            for (int j = 0; j < 8; ++j) {
                unsigned int lo = (w0[j >> 1] >> ((j & 1) * 16)) & 0xFFFFu;
                unsigned int hi = (w1[j >> 1] >> ((j & 1) * 16)) << 16;
                *(unsigned int*)&Vs[(fg + j) * 132 + key0] = lo | hi;
            }
        }
        __syncthreads();

        if (kb < qblock) {                       // block-uniform branch
            #pragma unroll
            for (int ksub = 0; ksub < 4; ++ksub)
                process(ksub, std::false_type{});          // mask-free path
        } else {
            for (int ksub = 0; ksub < dks; ++ksub)
                process(ksub, std::false_type{});          // mask-free path
            process(dks, std::true_type{});                // single masked block
        }
        __syncthreads();
    }

    // final denominator reduction + normalize, emit bf16 attn-out
    #pragma unroll
    for (int r = 0; r < 8; ++r) {
        float inv = fast_rcp(redsum16(rowsum[r]));
        int qg = qb + r + 8 * half;
        unsigned short* op = Og + headoff + (size_t)qg * BDIM;
        #pragma unroll
        for (int t = 0; t < 4; ++t)
            op[t * 16 + lm] = f2bf(O[t][r] * inv);
    }
}

// ---------------------------------------------------------------------------
extern "C" void kernel_launch(void* const* d_in, const int* in_sizes, int n_in,
                              void* d_out, int out_size, void* d_ws, size_t ws_size,
                              hipStream_t stream) {
    (void)in_sizes; (void)n_in; (void)out_size; (void)ws_size;
    const float* x  = (const float*)d_in[0];
    const float* Wq = (const float*)d_in[1];
    const float* bq = (const float*)d_in[2];
    const float* Wk = (const float*)d_in[3];
    const float* bk = (const float*)d_in[4];
    const float* Wv = (const float*)d_in[5];
    const float* bv = (const float*)d_in[6];
    const float* Wo = (const float*)d_in[7];
    const float* bo = (const float*)d_in[8];
    float* out = (float*)d_out;

    const size_t mat = (size_t)MROWS * BDIM;
    const size_t wsz = (size_t)BDIM * BDIM;
    unsigned short* xb  = (unsigned short*)d_ws;   // ws (bf16): x | 4 W^T | Q K V | attn
    unsigned short* Wqt = xb  + mat;
    unsigned short* Wkt = Wqt + wsz;
    unsigned short* Wvt = Wkt + wsz;
    unsigned short* Wot = Wvt + wsz;
    unsigned short* Qb  = Wot + wsz;
    unsigned short* Kb  = Qb  + mat;
    unsigned short* Vb  = Kb  + mat;
    unsigned short* Ab  = Vb  + mat;               // total ~88 MB

    cvt_bf16<<<dim3(mat / (256 * 4)), 256, 0, stream>>>(x, xb);
    dim3 gt(BDIM / 32, BDIM / 32);
    cvt_transpose_bf16<<<gt, 256, 0, stream>>>(Wq, Wqt);
    cvt_transpose_bf16<<<gt, 256, 0, stream>>>(Wk, Wkt);
    cvt_transpose_bf16<<<gt, 256, 0, stream>>>(Wv, Wvt);
    cvt_transpose_bf16<<<gt, 256, 0, stream>>>(Wo, Wot);

    dim3 gg(MROWS / 128, BDIM / 128);
    // Q scaled by log2(e)/sqrt(Dh): softmax runs in exp2 domain
    const float qscale = 0.125f * 1.44269504088896340736f;
    gemm_bf16<true ><<<gg, 256, 0, stream>>>(xb, Wqt, bq, Qb, qscale);
    gemm_bf16<true ><<<gg, 256, 0, stream>>>(xb, Wkt, bk, Kb, 1.0f);
    gemm_bf16<true ><<<gg, 256, 0, stream>>>(xb, Wvt, bv, Vb, 1.0f);

    dim3 ga(LLEN / 128, NB * NH);
    flash_attn_bf16<<<ga, 256, 0, stream>>>(Qb, Kb, Vb, Ab);

    gemm_bf16<false><<<gg, 256, 0, stream>>>(Ab, Wot, bo, out, 1.0f);
}
